// Attention_48069273977436
// MI455X (gfx1250) — compile-verified, matrix operand order checked by reading, not run on hardware
//
#include <hip/hip_runtime.h>
#include <hip/hip_bf16.h>

// ---------------------------------------------------------------------------
// Multi-head attention (B=8, N=1024, C=768, H=12, D=64) for gfx1250 (MI455X)
// All GEMMs via v_wmma_f32_16x16x32_f16 (f16 operands, f32 accumulate).
// Outputs: out [B,N,C] f32  followed by  attn [B,H,N,N] f32 in d_out.
// ---------------------------------------------------------------------------

typedef __attribute__((ext_vector_type(16))) _Float16 v16h;
typedef __attribute__((ext_vector_type(8)))  _Float16 v8h;
typedef __attribute__((ext_vector_type(8)))  float    v8f;

#define WMMA_F16(A, B, C) \
  __builtin_amdgcn_wmma_f32_16x16x32_f16(false, (A), false, (B), (short)0, (C), false, false)

#define BATCH 8
#define HEADS 12
#define NTOK  1024
#define DHEAD 64
#define CDIM  768
#define QKVO  2304          // 3*C
#define MROWS 8192          // B*N

// ---- A/B fragment from f16 row-major global memory ------------------------
// A (16x32 f16): lane = row (mod 16); K = {h*8+0..7, 16+h*8+0..7}, h=lane/16.
// B fragment for D = A * Bt^T is loaded from rows of Bt with the same pattern.
__device__ __forceinline__ v16h frag_ld_g(const _Float16* __restrict__ base,
                                          int row0, int k0, int ld) {
  const int lane = threadIdx.x & 31;
  const int hf = lane >> 4, r = lane & 15;
  const _Float16* p = base + (size_t)(row0 + r) * ld + k0 + hf * 8;
  v8h lo = *(const v8h*)p;
  v8h hi = *(const v8h*)(p + 16);
  v16h a;
#pragma unroll
  for (int i = 0; i < 8; ++i) { a[i] = lo[i]; a[i + 8] = hi[i]; }
  return a;
}

// ---- A fragment from f16 LDS strip (16 rows x 1024), scaled by f16 factor --
__device__ __forceinline__ v16h frag_ld_ph(const _Float16* __restrict__ P,
                                           int k0, _Float16 s) {
  const int lane = threadIdx.x & 31;
  const int hf = lane >> 4, r = lane & 15;
  const _Float16* p = P + r * NTOK + k0 + hf * 8;
  v8h lo = *(const v8h*)p;
  v8h hi = *(const v8h*)(p + 16);
  v16h a;
#pragma unroll
  for (int i = 0; i < 8; ++i) { a[i] = lo[i] * s; a[i + 8] = hi[i] * s; }
  return a;
}

// ---------------------------------------------------------------------------
__global__ void cvt_f32_to_f16(const float* __restrict__ in,
                               _Float16* __restrict__ out, int n) {
  int i = blockIdx.x * blockDim.x + threadIdx.x;
  if (i < n) out[i] = (_Float16)in[i];
}

// ---------------------------------------------------------------------------
// QKV = X[8192x768] * Wqkv^T[768x2304]; scatter into Q,K [B,H,N,D] and
// V transposed [B,H,D,N], all f16. One 32x16 tile per wave (2 WMMAs / k-step).
// ---------------------------------------------------------------------------
__global__ __launch_bounds__(128)
void qkv_gemm(const _Float16* __restrict__ xh,
              const _Float16* __restrict__ wq,
              _Float16* __restrict__ Qh,
              _Float16* __restrict__ Kh,
              _Float16* __restrict__ Vth) {
  const int wid = blockIdx.x * 4 + (threadIdx.x >> 5);   // 0 .. 256*144-1
  const int tm = wid % (MROWS / 32);                     // 0..255 (32-row tile)
  const int tn = wid / (MROWS / 32);                     // 0..143 (16-col tile)
  const int lane = threadIdx.x & 31;
  const int hf = lane >> 4, r = lane & 15;

  v8f c0 = {}, c1 = {};
  for (int k0 = 0; k0 < CDIM; k0 += 32) {
    v16h a0 = frag_ld_g(xh, tm * 32,      k0, CDIM);
    v16h a1 = frag_ld_g(xh, tm * 32 + 16, k0, CDIM);
    v16h b  = frag_ld_g(wq, tn * 16,      k0, CDIM);
    c0 = WMMA_F16(a0, b, c0);
    c1 = WMMA_F16(a1, b, c1);
  }

  const int o = tn * 16 + r;            // output column in [0,2304)
  const int s  = o / CDIM;              // 0=Q 1=K 2=V (uniform per tile)
  const int rm = o - s * CDIM;
  const int hh = rm / DHEAD;            // head (uniform per tile)
  const int dd = rm - hh * DHEAD;

  if (s == 2) {
    // V: each lane's 8 elements are consecutive tokens at fixed d in [B,H,D,N]
    // -> one 16-byte v8h store per half-tile (8-aligned, stays within batch).
#pragma unroll
    for (int half = 0; half < 2; ++half) {
      const v8f& c = half ? c1 : c0;
      v8h v;
#pragma unroll
      for (int i = 0; i < 8; ++i) v[i] = (_Float16)c[i];
      const int m0l = tm * 32 + half * 16 + 8 * hf;      // first token row
      const int bb = m0l >> 10, nn0 = m0l & (NTOK - 1);
      const size_t bhh = (size_t)(bb * HEADS + hh);
      *(v8h*)(Vth + (bhh * DHEAD + dd) * NTOK + nn0) = v;
    }
  } else {
#pragma unroll
    for (int half = 0; half < 2; ++half) {
      const v8f& c = half ? c1 : c0;
#pragma unroll
      for (int i = 0; i < 8; ++i) {
        const int m  = tm * 32 + half * 16 + i + 8 * hf; // token row [0,8192)
        const int bb = m >> 10, nn = m & (NTOK - 1);
        const _Float16 v = (_Float16)c[i];
        const size_t bhh = (size_t)(bb * HEADS + hh);
        if (s == 0) Qh[(bhh * NTOK + nn) * DHEAD + dd] = v;
        else        Kh[(bhh * NTOK + nn) * DHEAD + dd] = v;
      }
    }
  }
}

// ---------------------------------------------------------------------------
// Attention core: one wave per (b*h, 16-row strip).  32 KB f16 LDS strip ->
// 10 blocks/WGP.  S = Q K^T * scale (128 WMMAs) -> LDS f16; softmax with f32
// reductions over 16-byte LDS chunks (4 independent accumulators, bank-
// staggered); probs -> d_out as float4 stores; O = P V (128 WMMAs) reading
// f16 e-values straight from LDS.
// ---------------------------------------------------------------------------
__global__ __launch_bounds__(32)
void attn_core(const _Float16* __restrict__ Qh,
               const _Float16* __restrict__ Kh,
               const _Float16* __restrict__ Vth,
               _Float16* __restrict__ Oh,
               float* __restrict__ attnOut) {
  extern __shared__ _Float16 P[];              // 16 x 1024 f16 = 32 KB
  const int lane = threadIdx.x & 31;
  const int r = lane & 15, hf = lane >> 4;
  const int bh = blockIdx.y;                   // 0..95
  const int bb = bh / HEADS, hh = bh - bb * HEADS;
  const int m0 = blockIdx.x * 16;              // row strip

  const _Float16* Qb = Qh  + (size_t)bh * NTOK * DHEAD;
  const _Float16* Kb = Kh  + (size_t)bh * NTOK * DHEAD;
  const _Float16* Vb = Vth + (size_t)bh * DHEAD * NTOK;

  // --- scores: P[m0:m0+16, :] = scale * Q K^T (f16) --------------------------
  const v16h a0 = frag_ld_g(Qb, m0, 0, DHEAD);
  const v16h a1 = frag_ld_g(Qb, m0, 32, DHEAD);
  const float scale = 0.125f;                  // D^-0.5
  for (int t = 0; t < NTOK / 16; ++t) {
    v16h b0 = frag_ld_g(Kb, t * 16, 0, DHEAD);
    v16h b1 = frag_ld_g(Kb, t * 16, 32, DHEAD);
    v8f c = {};
    c = WMMA_F16(a0, b0, c);
    c = WMMA_F16(a1, b1, c);
#pragma unroll
    for (int i = 0; i < 8; ++i)
      P[(i + 8 * hf) * NTOK + t * 16 + r] = (_Float16)(c[i] * scale);
  }
  __syncthreads();

  // --- softmax: lane (r,hf) owns half-row segment (512) of row r -------------
  // Segment viewed as 64 x v8h chunks; start chunk staggered so all 32 lanes
  // hit distinct bank groups; 4 independent accumulators break the dep chain.
  v8h* Pv = (v8h*)(P + r * NTOK + hf * 512);
  const int stc = (2 * r + hf) & 63;
  float m0a = -3.4e38f, m1a = -3.4e38f, m2a = -3.4e38f, m3a = -3.4e38f;
  for (int cc = 0; cc < 64; cc += 4) {
    v8h x0 = Pv[(stc + cc + 0) & 63];
    v8h x1 = Pv[(stc + cc + 1) & 63];
    v8h x2 = Pv[(stc + cc + 2) & 63];
    v8h x3 = Pv[(stc + cc + 3) & 63];
#pragma unroll
    for (int k = 0; k < 8; ++k) {
      m0a = fmaxf(m0a, (float)x0[k]);
      m1a = fmaxf(m1a, (float)x1[k]);
      m2a = fmaxf(m2a, (float)x2[k]);
      m3a = fmaxf(m3a, (float)x3[k]);
    }
  }
  float mx = fmaxf(fmaxf(m0a, m1a), fmaxf(m2a, m3a));
  mx = fmaxf(mx, __shfl_xor(mx, 16, 32));

  float s0 = 0.f, s1 = 0.f, s2 = 0.f, s3 = 0.f;
  for (int cc = 0; cc < 64; cc += 4) {
    const int i0 = (stc + cc + 0) & 63, i1 = (stc + cc + 1) & 63;
    const int i2 = (stc + cc + 2) & 63, i3 = (stc + cc + 3) & 63;
    v8h x0 = Pv[i0], x1 = Pv[i1], x2 = Pv[i2], x3 = Pv[i3];
    v8h e0, e1, e2, e3;
#pragma unroll
    for (int k = 0; k < 8; ++k) {
      float e;
      e = __expf((float)x0[k] - mx); e0[k] = (_Float16)e; s0 += e;
      e = __expf((float)x1[k] - mx); e1[k] = (_Float16)e; s1 += e;
      e = __expf((float)x2[k] - mx); e2[k] = (_Float16)e; s2 += e;
      e = __expf((float)x3[k] - mx); e3[k] = (_Float16)e; s3 += e;
    }
    Pv[i0] = e0; Pv[i1] = e1; Pv[i2] = e2; Pv[i3] = e3;
  }
  float sum = (s0 + s1) + (s2 + s3);
  sum += __shfl_xor(sum, 16, 32);
  const float inv = 1.f / sum;                 // lane holds inv of row (lane&15)
  __syncthreads();

  // --- write softmax probabilities: float4 stores, coalesced per row ---------
  float* abase = attnOut + ((size_t)bh * NTOK + m0) * NTOK;
  for (int i = 0; i < 16; ++i) {
    const float invi = __shfl(inv, i, 32);     // lane i holds row i's inv
    const _Float16* src = P + i * NTOK;
    float* dst = abase + (size_t)i * NTOK;
    for (int j0 = lane * 4; j0 < NTOK; j0 += 128) {
      float4 v;
      v.x = (float)src[j0 + 0] * invi;
      v.y = (float)src[j0 + 1] * invi;
      v.z = (float)src[j0 + 2] * invi;
      v.w = (float)src[j0 + 3] * invi;
      *(float4*)(dst + j0) = v;
    }
  }

  // --- O = P V : A-fragments straight from f16 LDS, scaled by f16 1/sum ------
  const _Float16 hinv = (_Float16)inv;
  v8f c0 = {}, c1 = {}, c2 = {}, c3 = {};
  for (int kt = 0; kt < NTOK; kt += 32) {
    v16h ap = frag_ld_ph(P, kt, hinv);
    v16h b0 = frag_ld_g(Vb, 0,  kt, NTOK);
    v16h b1 = frag_ld_g(Vb, 16, kt, NTOK);
    v16h b2 = frag_ld_g(Vb, 32, kt, NTOK);
    v16h b3 = frag_ld_g(Vb, 48, kt, NTOK);
    c0 = WMMA_F16(ap, b0, c0);
    c1 = WMMA_F16(ap, b1, c1);
    c2 = WMMA_F16(ap, b2, c2);
    c3 = WMMA_F16(ap, b3, c3);
  }
#pragma unroll
  for (int i = 0; i < 8; ++i) {
    const int n = m0 + i + 8 * hf;
    const size_t ro = ((size_t)bb * NTOK + n) * CDIM + hh * DHEAD;
    Oh[ro + 0 * 16 + r] = (_Float16)c0[i];
    Oh[ro + 1 * 16 + r] = (_Float16)c1[i];
    Oh[ro + 2 * 16 + r] = (_Float16)c2[i];
    Oh[ro + 3 * 16 + r] = (_Float16)c3[i];
  }
}

// ---------------------------------------------------------------------------
// out = Ctx[8192x768] * Wproj^T + bias, f32 result.  32x16 tile per wave.
// ---------------------------------------------------------------------------
__global__ __launch_bounds__(128)
void proj_gemm(const _Float16* __restrict__ oh,
               const _Float16* __restrict__ wp,
               const float* __restrict__ bias,
               float* __restrict__ out) {
  const int wid = blockIdx.x * 4 + (threadIdx.x >> 5);  // 0 .. 256*48-1
  const int tm = wid % (MROWS / 32);
  const int tn = wid / (MROWS / 32);
  const int lane = threadIdx.x & 31;
  const int hf = lane >> 4, r = lane & 15;

  v8f c0 = {}, c1 = {};
  for (int k0 = 0; k0 < CDIM; k0 += 32) {
    v16h a0 = frag_ld_g(oh, tm * 32,      k0, CDIM);
    v16h a1 = frag_ld_g(oh, tm * 32 + 16, k0, CDIM);
    v16h b  = frag_ld_g(wp, tn * 16,      k0, CDIM);
    c0 = WMMA_F16(a0, b, c0);
    c1 = WMMA_F16(a1, b, c1);
  }
  const int o = tn * 16 + r;
  const float bo = bias[o];
#pragma unroll
  for (int half = 0; half < 2; ++half) {
    const v8f& c = half ? c1 : c0;
#pragma unroll
    for (int i = 0; i < 8; ++i) {
      const int m = tm * 32 + half * 16 + i + 8 * hf;
      out[(size_t)m * CDIM + o] = c[i] + bo;
    }
  }
}

// ---------------------------------------------------------------------------
extern "C" void kernel_launch(void* const* d_in, const int* in_sizes, int n_in,
                              void* d_out, int out_size, void* d_ws, size_t ws_size,
                              hipStream_t stream) {
  const float* x      = (const float*)d_in[0];   // [8,1024,768]
  const float* qkv_w  = (const float*)d_in[1];   // [2304,768]
  const float* proj_w = (const float*)d_in[2];   // [768,768]
  const float* proj_b = (const float*)d_in[3];   // [768]

  float* out  = (float*)d_out;                          // [8,1024,768]
  float* attn = out + (size_t)MROWS * CDIM;             // [8,12,1024,1024]

  // f16 workspace carve-out (~67.6 MB total)
  _Float16* xh   = (_Float16*)d_ws;
  _Float16* wqh  = xh  + (size_t)MROWS * CDIM;          // 6291456
  _Float16* wph  = wqh + (size_t)QKVO * CDIM;           // 1769472
  _Float16* Qh   = wph + (size_t)CDIM * CDIM;           // 589824
  _Float16* Kh   = Qh  + (size_t)6291456;
  _Float16* Vth  = Kh  + (size_t)6291456;
  _Float16* Oh   = Vth + (size_t)6291456;

  const int nx = MROWS * CDIM;          // 6291456
  const int nq = QKVO * CDIM;           // 1769472
  const int np = CDIM * CDIM;           // 589824
  cvt_f32_to_f16<<<(nx + 255) / 256, 256, 0, stream>>>(x, xh, nx);
  cvt_f32_to_f16<<<(nq + 255) / 256, 256, 0, stream>>>(qkv_w, wqh, nq);
  cvt_f32_to_f16<<<(np + 255) / 256, 256, 0, stream>>>(proj_w, wph, np);

  // 256 * 144 tiles / 4 waves per block
  qkv_gemm<<<(256 * 144) / 4, 128, 0, stream>>>(xh, wqh, Qh, Kh, Vth);

  // one wave per (b*h, 16-row strip), 32 KB dynamic LDS
  attn_core<<<dim3(NTOK / 16, BATCH * HEADS), 32, 16 * NTOK * sizeof(_Float16), stream>>>(
      Qh, Kh, Vth, Oh, attn);

  // 256 * 48 tiles / 4 waves per block
  proj_gemm<<<(256 * 48) / 4, 128, 0, stream>>>(Oh, wph, proj_b, out);
}
